// LinearQ4_0_52596169506909
// MI455X (gfx1250) — compile-verified, matrix operand order checked by reading, not run on hardware
//
#include <hip/hip_runtime.h>
#include <stdint.h>

typedef __attribute__((ext_vector_type(16))) _Float16 v16h;
typedef __attribute__((ext_vector_type(8)))  _Float16 v8h;
typedef __attribute__((ext_vector_type(2)))  _Float16 h2;
typedef __attribute__((ext_vector_type(8)))  float    v8f;
typedef __attribute__((ext_vector_type(4)))  int      v4i;

#define AS1 __attribute__((address_space(1)))
#define AS3 __attribute__((address_space(3)))

namespace {
constexpr int K_DIM  = 4096;
constexpr int N_DIM  = 11008;
constexpr int BM     = 128;
constexpr int BN     = 128;
constexpr int BK     = 32;
constexpr int LDA    = BK + 8;          // 80B LDS row stride: conflict-free b128 access
constexpr int GROUP  = 64;
constexpr int KSTEPS = K_DIM / BK;      // 128
}

// ---- gfx1250 async global->LDS copy (ASYNCcnt-tracked, bypasses VGPRs) ----
__device__ __forceinline__ void async_b128(const void* g, void* l) {
#if __has_builtin(__builtin_amdgcn_global_load_async_to_lds_b128)
    __builtin_amdgcn_global_load_async_to_lds_b128(
        (AS1 v4i*)(void*)g, (AS3 v4i*)l, 0, 0);
#else
    const uint32_t loff = (uint32_t)(uintptr_t)(AS3 void*)l;
    asm volatile("global_load_async_to_lds_b128 %0, %1, off"
                 :: "v"(loff), "v"(g) : "memory");
#endif
}

__device__ __forceinline__ void wait_async0() {
#if __has_builtin(__builtin_amdgcn_s_wait_asynccnt)
    __builtin_amdgcn_s_wait_asynccnt(0);
#else
    asm volatile("s_wait_asynccnt 0x0" ::: "memory");
#endif
}

// ---- bit-exact packed int8 -> fp16*scale dequant ----
// q^0x80 -> 0x6400|u is fp16 value (1152 + q) exactly; subtract 1152, mul scale.
// 7 VALU per 4 weights (xor + 2 perm + 2 pk_add + 2 pk_mul).
__device__ __forceinline__ void dequant16(const uint32_t qw[4], _Float16 s,
                                          v8h& lo, v8h& hi) {
    const h2 s2 = {s, s};
    const h2 c  = {(_Float16)1152.0f, (_Float16)1152.0f};
    union V8 { v8h v; h2 h[4]; } L, H;
    #pragma unroll
    for (int d = 0; d < 4; ++d) {
        const uint32_t q  = qw[d] ^ 0x80808080u;
        const uint32_t p0 = __builtin_amdgcn_perm(0x64646464u, q, 0x04010400u);
        const uint32_t p1 = __builtin_amdgcn_perm(0x64646464u, q, 0x04030402u);
        const h2 f0 = (__builtin_bit_cast(h2, p0) - c) * s2;
        const h2 f1 = (__builtin_bit_cast(h2, p1) - c) * s2;
        V8& dst = (d < 2) ? L : H;
        dst.h[(d & 1) * 2 + 0] = f0;
        dst.h[(d & 1) * 2 + 1] = f1;
    }
    lo = L.v; hi = H.v;
}

__global__ __launch_bounds__(256)
void linearq8_wmma_db(const _Float16* __restrict__ x,
                      const int8_t*   __restrict__ wq,
                      const _Float16* __restrict__ scale,
                      _Float16*       __restrict__ out)
{
    __shared__ __align__(16) _Float16 sA[2][BM * LDA];  // activations, fp16
    __shared__ __align__(16) _Float16 sB[2][BN * LDA];  // dequantized weights

    const int tid  = threadIdx.x;
    const int lane = tid & 31;
    const int wave = tid >> 5;
    const int wm   = wave & 1;          // 64-row M slice
    const int wn   = wave >> 1;         // 32-col N slice
    const int m0   = blockIdx.y * BM;
    const int n0   = blockIdx.x * BN;
    const int bRow = tid >> 1;          // W row in tile (0..127)
    const int bK   = (tid & 1) * 16;    // int8 k offset 0/16
    const int fRow = lane & 15;
    const int fKo  = (lane >> 4) * 8;

    const _Float16* aBase = x + (size_t)m0 * K_DIM;
    const int8_t*   bBase = wq + (size_t)(n0 + bRow) * K_DIM + bK;
    const _Float16* sBase = scale + (size_t)(n0 + bRow) * (K_DIM / GROUP);

    const v8f zero = {0.f, 0.f, 0.f, 0.f, 0.f, 0.f, 0.f, 0.f};
    v8f acc[4][2];
    #pragma unroll
    for (int i = 0; i < 4; ++i)
        #pragma unroll
        for (int j = 0; j < 2; ++j)
            acc[i][j] = zero;

    // --- pipeline stages ---
    auto stageA = [&](int buf, int k0) {        // async: 2 x b128 per thread
        #pragma unroll
        for (int c = 0; c < 2; ++c) {
            const int chunk = tid + c * 256;    // 512 x 16B chunks
            const int row   = chunk >> 2;
            const int ko    = (chunk & 3) * 8;
            async_b128(&aBase[(size_t)row * K_DIM + k0 + ko],
                       &sA[buf][row * LDA + ko]);
        }
    };

    uint4    qreg;                              // in-flight B bytes
    _Float16 sreg;                              // in-flight group scale
    auto loadB = [&](int k0) {
        qreg = *reinterpret_cast<const uint4*>(bBase + k0);
        sreg = sBase[(k0 + bK) >> 6];           // one Q8_0 group per 16B chunk
    };
    auto commitB = [&](int buf) {
        const uint32_t qw[4] = {qreg.x, qreg.y, qreg.z, qreg.w};
        v8h lo, hi;
        dequant16(qw, sreg, lo, hi);
        *reinterpret_cast<v8h*>(&sB[buf][bRow * LDA + bK])     = lo;
        *reinterpret_cast<v8h*>(&sB[buf][bRow * LDA + bK + 8]) = hi;
    };

    auto compute = [&](int buf) {
        v16h a[4], b[2];
        #pragma unroll
        for (int i = 0; i < 4; ++i) {
            const int base = (wm * 64 + i * 16 + fRow) * LDA + fKo;
            v8h lo = *reinterpret_cast<const v8h*>(&sA[buf][base]);
            v8h hi = *reinterpret_cast<const v8h*>(&sA[buf][base + 16]);
            #pragma unroll
            for (int t = 0; t < 8; ++t) { a[i][t] = lo[t]; a[i][t + 8] = hi[t]; }
        }
        #pragma unroll
        for (int j = 0; j < 2; ++j) {
            const int base = (wn * 32 + j * 16 + fRow) * LDA + fKo;
            v8h lo = *reinterpret_cast<const v8h*>(&sB[buf][base]);
            v8h hi = *reinterpret_cast<const v8h*>(&sB[buf][base + 16]);
            #pragma unroll
            for (int t = 0; t < 8; ++t) { b[j][t] = lo[t]; b[j][t + 8] = hi[t]; }
        }
        #pragma unroll
        for (int i = 0; i < 4; ++i)
            #pragma unroll
            for (int j = 0; j < 2; ++j)
                acc[i][j] = __builtin_amdgcn_wmma_f32_16x16x32_f16(
                    false, a[i], false, b[j], (short)0, acc[i][j], false, false);
    };

    // --- prologue: stage tile 0 ---
    stageA(0, 0);
    loadB(0);
    commitB(0);
    wait_async0();
    __syncthreads();

    // --- double-buffered main loop ---
    for (int t = 0; t < KSTEPS; ++t) {
        const int  cur     = t & 1;
        const int  nxt     = cur ^ 1;
        const bool hasNext = (t + 1) < KSTEPS;
        if (hasNext) {
            stageA(nxt, (t + 1) * BK);          // async A -> LDS[nxt]
            loadB((t + 1) * BK);                // B bytes -> regs (in flight)
        }
        compute(cur);                           // 8 WMMAs from LDS[cur]
        if (hasNext) commitB(nxt);              // dequant + ds_store -> LDS[nxt]
        wait_async0();                          // A tile landed in LDS[nxt]
        __syncthreads();                        // orders reads(cur) vs writes(nxt)
    }

    // --- epilogue: C/D layout -> N = lane&15, M = vgpr + 8*(lane>>4) ---
    #pragma unroll
    for (int i = 0; i < 4; ++i) {
        #pragma unroll
        for (int j = 0; j < 2; ++j) {
            const int gm = m0 + wm * 64 + i * 16 + (lane >> 4) * 8;
            const int gn = n0 + wn * 32 + j * 16 + (lane & 15);
            #pragma unroll
            for (int v = 0; v < 8; ++v)
                out[(size_t)(gm + v) * N_DIM + gn] = (_Float16)acc[i][j][v];
        }
    }
}

extern "C" void kernel_launch(void* const* d_in, const int* in_sizes, int n_in,
                              void* d_out, int out_size, void* d_ws, size_t ws_size,
                              hipStream_t stream) {
    const _Float16* x   = (const _Float16*)d_in[0];   // fp16 [4,2048,4096]
    const int8_t*   w   = (const int8_t*)d_in[1];     // int8 flat [N*K]
    const _Float16* scl = (const _Float16*)d_in[2];   // fp16 per-group scales
    _Float16*       out = (_Float16*)d_out;           // fp16 [4,2048,11008]

    const int M = in_sizes[0] / K_DIM;                // 8192
    dim3 grid(N_DIM / BN, M / BM);                    // 86 x 64
    linearq8_wmma_db<<<grid, dim3(256), 0, stream>>>(x, w, scl, out);

    (void)n_in; (void)out_size; (void)d_ws; (void)ws_size;
}
